// PCNNNet_7172595385049
// MI455X (gfx1250) — matrix operand order repaired and checked by reading, not compiled, not run on hardware
//
#include <hip/hip_runtime.h>

// ---- CDNA5 (gfx1250) types for WMMA ----
typedef __bf16 v16bf __attribute__((ext_vector_type(16)));
typedef __bf16 v8bf  __attribute__((ext_vector_type(8)));
typedef __bf16 v4bf  __attribute__((ext_vector_type(4)));
typedef float  v8f   __attribute__((ext_vector_type(8)));

#define SEQ    40
#define EMBD   300
#define EMBP   328   // LDS row stride (elems): 328*2B -> 164 dw stride, spreads 64 banks
#define KTAPP  320   // padded K per conv tap (10 chunks of 32)
#define NCHN   210
#define NTILE  14    // ceil(210/16) n-tiles (padded to 224)
#define NTPW   7     // n-tiles per wave (N split across 2 waves)
#define KCH    30    // 3 taps * 10 k-chunks (even, required by 2x unroll)
#define BPB    4     // batches per block
#define ROWS   160   // BPB*SEQ  (M tile)
#define MSUBS  10    // 16-row M-subtiles per block
#define THREADS 640  // 20 waves: wave = msub + 10*nhalf
#define CSTRIDE 212  // fp32 conv-out LDS stride
#define NBLK   5
#define KPOOL  38
#define DH     35
#define NCLS   35
#define BCHUNK (NTILE*32*16)   // packed-B bf16 elems per k-chunk = 7168

struct P1 {
  __bf16 emb[161 * EMBP];      // rows 0..159 = emb (bf16), row 160 = zeros (conv padding)
};
struct P2 {
  float c[ROWS * CSTRIDE];     // conv output (post bias+ReLU), fp32
  float colmax[BPB * NBLK * SEQ];
  float h[BPB * NBLK * DH];
  int ep1[BPB];
  int ep2[BPB];
};
union SMem { P1 p1; P2 p2; };

// ---------------- Kernel 0: pack conv_w -> bf16 WMMA-B fragment layout ----------------
// Layout index: [kchunk][ntile][lane][i]  (lane: n = lane&15, h=lane>>4; elem i -> K = i<8 ? 8h+i : 16+8h+(i-8))
__global__ void pcnn_pack_w(const float* __restrict__ conv_w, __bf16* __restrict__ bpack) {
  int idx = blockIdx.x * 256 + threadIdx.x;
  const int total = KCH * BCHUNK;
  if (idx >= total) return;
  int i    = idx & 15;
  int lane = (idx >> 4) & 31;
  int t    = (idx >> 9) % NTILE;
  int kc   = idx / (NTILE * 512);
  int tap  = kc / 10;
  int kk   = (kc % 10) * 32 + ((lane >> 4) * 8) + (i < 8 ? i : 16 + (i - 8));   // K within tap (padded to 320)
  int n    = t * 16 + (lane & 15);                      // channel (padded to 224)
  float v  = (kk < EMBD && n < NCHN) ? conv_w[(tap * EMBD + kk) * NCHN + n] : 0.0f;
  bpack[idx] = (__bf16)v;
}

// coalesced 32B/lane B-fragment load straight from the pre-swizzled global buffer
__device__ __forceinline__ v16bf load_bfrag(const __bf16* __restrict__ bpack,
                                            int kc, int t, int lane) {
  return *(const v16bf*)(bpack + (size_t)kc * BCHUNK + ((size_t)(t * 32 + lane) << 4));
}

__device__ __forceinline__ void load_chunk(v16bf buf[NTPW], const __bf16* __restrict__ bpack,
                                           int kc, int nh, int lane) {
#pragma unroll
  for (int t = 0; t < NTPW; ++t) buf[t] = load_bfrag(bpack, kc, nh * NTPW + t, lane);
}

// A fragment from LDS (16-bit A 16x32 layout): lo 16B at (half*16), hi 16B at +32B
__device__ __forceinline__ v16bf load_afrag(const __bf16* __restrict__ emb,
                                            int kc, int bb, int s, int half) {
  const int tap  = kc / 10;
  const int koff = (kc % 10) * 32;
  const int ss   = s + tap - 1;                                   // conv tap shift
  const int srow = ((unsigned)ss < SEQ) ? (bb * SEQ + ss) : 160;  // 160 = zero row
  const __bf16* abase = emb + srow * EMBP + koff + half * 8;
  v8bf alo = *(const v8bf*)abase;
  v8bf ahi = *(const v8bf*)(abase + 16);
  v16bf a;
#pragma unroll
  for (int i = 0; i < 8; ++i) { a[i] = alo[i]; a[8 + i] = ahi[i]; }
  return a;
}

__device__ __forceinline__ void wmma7(v8f acc[NTPW], v16bf a, const v16bf buf[NTPW]) {
#pragma unroll
  for (int t = 0; t < NTPW; ++t) {
    acc[t] = __builtin_amdgcn_wmma_f32_16x16x32_bf16(
        /*neg_a=*/false, a, /*neg_b=*/false, buf[t],
        /*c_mod=*/(short)0, acc[t], /*reuse_a=*/false, /*reuse_b=*/false);
  }
}

// ---------------- Kernel 1: fused embed -> conv GEMM (WMMA) -> piecewise pool -> MLP ----------------
__global__ __launch_bounds__(THREADS, 1) void pcnn_main(
    const int*   __restrict__ x,       const int*   __restrict__ pos_,
    const float* __restrict__ w2v,     const float* __restrict__ conv_b,
    const float* __restrict__ dense_w, const float* __restrict__ dense_b,
    const float* __restrict__ out_w,   const float* __restrict__ out_b,
    const __bf16* __restrict__ bpack,  float* __restrict__ out)
{
  __shared__ SMem sm;
  const int tid  = threadIdx.x;
  const int lane = tid & 31;
  const int wave = tid >> 5;             // 0..19
  const int msub = wave % MSUBS;         // 16-row M-subtile
  const int nh   = wave / MSUBS;         // N-half: tiles [nh*7, nh*7+7)
  const int b0   = blockIdx.x * BPB;     // first batch of this block

  // ---- Stage embeddings for 4 batches: gather from w2v (L2-resident), cvt fp32->bf16 ----
  for (int q = tid; q < ROWS * (EMBD / 4); q += THREADS) {
    int r  = q / (EMBD / 4);
    int cc = q % (EMBD / 4);
    int tok = x[b0 * SEQ + r];
    const float4 v = ((const float4*)(w2v + (size_t)tok * EMBD))[cc];
    v4bf o; o[0] = (__bf16)v.x; o[1] = (__bf16)v.y; o[2] = (__bf16)v.z; o[3] = (__bf16)v.w;
    *(v4bf*)&sm.p1.emb[r * EMBP + cc * 4] = o;
  }
  for (int q = tid; q < ROWS * (KTAPP - EMBD); q += THREADS) {          // zero K-pad cols
    int r = q / (KTAPP - EMBD), j = q % (KTAPP - EMBD);
    sm.p1.emb[r * EMBP + EMBD + j] = (__bf16)0.0f;
  }
  for (int q = tid; q < KTAPP; q += THREADS)                            // zero padding row
    sm.p1.emb[160 * EMBP + q] = (__bf16)0.0f;
  __syncthreads();

  // ---- Main GEMM: M=160 x N=224 x K=960, bf16 WMMA 16x16x32; each wave: 16 rows x 7 tiles ----
  v8f acc[NTPW];
#pragma unroll
  for (int t = 0; t < NTPW; ++t) {
#pragma unroll
    for (int i = 0; i < 8; ++i) acc[t][i] = 0.0f;
  }

  const int m    = lane & 15;
  const int half = lane >> 4;
  const int r    = msub * 16 + m;        // row of this lane's A fragment
  const int bb   = r / SEQ, s = r % SEQ;

  // chunk-granular double buffering: WMMAs on bufA overlap the in-flight loads of bufB.
  // Live ranges overlap by construction -> distinct registers, partial loadcnt waits.
  v16bf bufA[NTPW], bufB[NTPW];
  load_chunk(bufA, bpack, 0, nh, lane);

  for (int kc = 0; kc < KCH; kc += 2) {
    const int kc2 = (kc + 2 < KCH) ? kc + 2 : 0;   // wrapped dummy on the tail

    v16bf a0 = load_afrag(sm.p1.emb, kc, bb, s, half);
    load_chunk(bufB, bpack, kc + 1, nh, lane);     // stream next chunk while computing
    __builtin_prefetch(bpack + (size_t)kc2 * BCHUNK + (size_t)lane * (BCHUNK / 32), 0, 3);
    wmma7(acc, a0, bufA);

    v16bf a1 = load_afrag(sm.p1.emb, kc + 1, bb, s, half);
    load_chunk(bufA, bpack, kc2, nh, lane);        // stream chunk kc+2 while computing
    wmma7(acc, a1, bufB);
  }

  // ---- Epilogue: bias + ReLU, spill conv-out to LDS (union reuse) ----
  __syncthreads();
#pragma unroll
  for (int t = 0; t < NTPW; ++t) {
    int n = (nh * NTPW + t) * 16 + (lane & 15);
    float bias = (n < NCHN) ? conv_b[n] : 0.0f;
#pragma unroll
    for (int v = 0; v < 8; ++v) {
      int rr = msub * 16 + v + half * 8;      // C/D layout: M = v + 8*(lane>>4), N = lane&15
      float val = acc[t][v] + bias;
      val = val > 0.0f ? val : 0.0f;
      if (n < NCHN) sm.p2.c[rr * CSTRIDE + n] = val;
    }
  }
  if (tid < BPB) {    // entity positions -> piecewise boundaries
    int p0 = pos_[(b0 + tid) * 2 + 0];
    int p1 = pos_[(b0 + tid) * 2 + 1];
    int e1 = p0 < p1 ? p0 : p1;
    int e2 = p0 < p1 ? p1 : p0;
    int sh = (e1 == 0) ? 1 : 0;
    sm.p2.ep1[tid] = e1 + sh;
    sm.p2.ep2[tid] = e2 + sh;
  }
  __syncthreads();

  // ---- channel-block max over 38 channels: colmax[bb][nb][s] ----
  for (int idx = tid; idx < BPB * NBLK * SEQ; idx += THREADS) {
    int bbq = idx / (NBLK * SEQ);
    int nb  = (idx / SEQ) % NBLK;
    int sq  = idx % SEQ;
    const float* col = &sm.p2.c[(bbq * SEQ + sq) * CSTRIDE + nb * KPOOL];
    float mx = col[0];
    for (int chn = 1; chn < KPOOL; ++chn) mx = fmaxf(mx, col[chn]);
    sm.p2.colmax[idx] = mx;
  }
  __syncthreads();

  // ---- dense: [20,120] @ [120,35] + b, with piecewise mask applied on the fly ----
  for (int o = tid; o < BPB * NBLK * DH; o += THREADS) {
    int row = o / DH, j = o % DH;
    int bbq = row / NBLK;
    int e1 = sm.p2.ep1[bbq], e2 = sm.p2.ep2[bbq];
    float ad = dense_b[j];
    for (int k = 0; k < 3 * SEQ; ++k) {
      int rgn = k / SEQ, sq = k % SEQ;
      bool msk = (rgn == 0) ? (sq < e1)
               : (rgn == 1) ? (sq >= e1 && sq < e2)
                            : (sq >= e2);
      float pv = msk ? sm.p2.colmax[row * SEQ + sq] : 0.0f;
      ad += pv * dense_w[k * DH + j];
    }
    sm.p2.h[o] = ad;
  }
  __syncthreads();

  // ---- out: [20,35] @ [35,35] + b -> global ----
  for (int o = tid; o < BPB * NBLK * NCLS; o += THREADS) {
    int row = o / NCLS, j2 = o % NCLS;
    float ao = out_b[j2];
    for (int j = 0; j < DH; ++j) ao += sm.p2.h[row * DH + j] * out_w[j * NCLS + j2];
    int bg = b0 + row / NBLK, nb = row % NBLK;
    out[(size_t)(bg * NBLK + nb) * NCLS + j2] = ao;
  }
}

// ---------------- host ----------------
extern "C" void kernel_launch(void* const* d_in, const int* in_sizes, int n_in,
                              void* d_out, int out_size, void* d_ws, size_t ws_size,
                              hipStream_t stream) {
  const int*   x      = (const int*)d_in[0];
  const int*   pos    = (const int*)d_in[1];
  const float* w2v    = (const float*)d_in[2];
  const float* conv_w = (const float*)d_in[3];
  const float* conv_b = (const float*)d_in[4];
  const float* dw     = (const float*)d_in[5];
  const float* db     = (const float*)d_in[6];
  const float* ow     = (const float*)d_in[7];
  const float* ob     = (const float*)d_in[8];
  float* out = (float*)d_out;
  __bf16* bpack = (__bf16*)d_ws;               // 430,080 B of scratch

  const int total = KCH * BCHUNK;              // 215,040 bf16 elems
  pcnn_pack_w<<<(total + 255) / 256, 256, 0, stream>>>(conv_w, bpack);
  pcnn_main<<<8192 / BPB, THREADS, 0, stream>>>(x, pos, w2v, conv_b, dw, db, ow, ob, bpack, out);
}
